// BinarizedConv2d_90099823935570
// MI455X (gfx1250) — 8-bit matrix operand order repaired and checked by reading, compiled, not run on hardware
//
#include <hip/hip_runtime.h>

typedef __attribute__((ext_vector_type(8))) int v8i;

#define NIMG 32
#define CIN 256
#define COUT 256
#define HW 56
#define HP 58
#define SPATIAL (HW * HW)
#define XPAD_INTS (NIMG * HP * HP * (CIN/4))
#define XPAD_BYTES (NIMG * HP * HP * CIN)
#define WQ_BYTES (9 * COUT * CIN)

__global__ void zero_ints(int* __restrict__ p, int n) {
    int i = blockIdx.x * blockDim.x + threadIdx.x;
    if (i < n) p[i] = 0;
}

__device__ __forceinline__ int sgn8(float v) { return v > 0.f ? 1 : (v < 0.f ? -1 : 0); }

__global__ void binarize_x_kernel(const float* __restrict__ x, int* __restrict__ xpad) {
    int t = blockIdx.x * blockDim.x + threadIdx.x;
    int c4 = t & 63;
    int r = t >> 6;
    int xw = r % HW; r /= HW;
    int yh = r % HW;
    int n  = r / HW;
    size_t ib = (((size_t)n * CIN + c4 * 4) * HW + yh) * HW + xw;
    int b0 = sgn8(x[ib]);
    int b1 = sgn8(x[ib + SPATIAL]);
    int b2 = sgn8(x[ib + 2 * SPATIAL]);
    int b3 = sgn8(x[ib + 3 * SPATIAL]);
    int packed = (b0 & 0xff) | ((b1 & 0xff) << 8) | ((b2 & 0xff) << 16) | ((b3 & 0xff) << 24);
    xpad[(((size_t)n * HP + yh + 1) * HP + (xw + 1)) * (CIN / 4) + c4] = packed;
}

__global__ void binarize_w_kernel(const float* __restrict__ w, signed char* __restrict__ wq) {
    int t = blockIdx.x * blockDim.x + threadIdx.x;
    int ic = t % CIN;
    int r = t / CIN;
    int oc = r % COUT;
    int tap = r / COUT;
    float v = w[((size_t)oc * CIN + ic) * 9 + tap];
    wq[(size_t)tap * (COUT * CIN) + (size_t)oc * CIN + ic] = (signed char)sgn8(v);
}

__global__ __launch_bounds__(256) void conv_bnn_wmma(
    const signed char* __restrict__ xpad,
    const signed char* __restrict__ wq,
    const float* __restrict__ bias,
    float* __restrict__ out)
{
    const int lane = threadIdx.x & 31;
    const int wave = threadIdx.x >> 5;
    const int wm   = wave & 3;
    const int wn   = wave >> 2;
    const int col  = lane & 15;
    const int half = lane >> 4;

    const int pbase = blockIdx.x * 128 + wn * 64;

    const signed char* colbase[4];
    int nn[4], ss[4];
#pragma unroll
    for (int nt = 0; nt < 4; ++nt) {
        int p = pbase + nt * 16 + col;
        int n = p / SPATIAL;
        int s = p % SPATIAL;
        nn[nt] = n; ss[nt] = s;
        int yy = s / HW, xx = s % HW;
        colbase[nt] = xpad + (((size_t)n * HP + yy) * HP + xx) * CIN;
    }

    v8i acc[4][4] = {};

    for (int tap = 0; tap < 9; ++tap) {
        const int ky = tap / 3, kx = tap % 3;
        const int coff = (ky * HP + kx) * CIN;
        const signed char* wb = wq + (size_t)tap * (COUT * CIN);

        if (tap < 8) {
            const int nky = (tap + 1) / 3, nkx = (tap + 1) % 3;
            const int ncoff = (nky * HP + nkx) * CIN;
#pragma unroll
            for (int nt = 0; nt < 4; ++nt)
                __builtin_prefetch(colbase[nt] + ncoff, 0, 1);
        }

#pragma unroll
        for (int kc = 0; kc < 4; ++kc) {
            v8i B[4];
#pragma unroll
            for (int nt = 0; nt < 4; ++nt) {
                const int4* bp = (const int4*)(colbase[nt] + coff + kc * 64 + half * 16);
                int4 blo = bp[0];
                int4 bhi = bp[2];
                v8i tb = { blo.x, blo.y, blo.z, blo.w, bhi.x, bhi.y, bhi.z, bhi.w };
                B[nt] = tb;
            }
            v8i A[4];
#pragma unroll
            for (int mt = 0; mt < 4; ++mt) {
                const int2* ap = (const int2*)(wb +
                    (size_t)(wm * 64 + mt * 16 + col) * CIN + kc * 64 + half * 16);
                int2 a0 = ap[0], a1 = ap[1], a2 = ap[4], a3 = ap[5];
                v8i ta = { a0.x, a0.y, a1.x, a1.y, a2.x, a2.y, a3.x, a3.y };
                A[mt] = ta;
            }
#pragma unroll
            for (int mt = 0; mt < 4; ++mt) {
#pragma unroll
                for (int nt = 0; nt < 4; ++nt) {
                    acc[mt][nt] = __builtin_amdgcn_wmma_i32_16x16x64_iu8(
                         true, A[mt],  true, B[nt], acc[mt][nt],
                         false,  false);
                }
            }
        }
    }

#pragma unroll
    for (int nt = 0; nt < 4; ++nt) {
        float* obase = out + (size_t)nn[nt] * COUT * SPATIAL + ss[nt];
#pragma unroll
        for (int mt = 0; mt < 4; ++mt) {
#pragma unroll
            for (int r = 0; r < 8; ++r) {
                int oc = wm * 64 + mt * 16 + half * 8 + r;
                obase[(size_t)oc * SPATIAL] = (float)acc[mt][nt][r] + bias[oc];
            }
        }
    }
}

__global__ void channel_reduce(const float* __restrict__ out,
                               float* __restrict__ sums, float* __restrict__ sumsq) {
    __shared__ float sh1[256];
    __shared__ float sh2[256];
    int c = blockIdx.x;
    int chunk = blockIdx.y;
    int t = threadIdx.x;
    float s1 = 0.f, s2 = 0.f;
    for (int i = t; i < 4 * SPATIAL; i += 256) {
        int n = chunk * 4 + i / SPATIAL;
        int s = i % SPATIAL;
        float v = out[((size_t)n * COUT + c) * SPATIAL + s];
        s1 += v;
        s2 += v * v;
    }
    sh1[t] = s1; sh2[t] = s2;
    __syncthreads();
    for (int off = 128; off > 0; off >>= 1) {
        if (t < off) { sh1[t] += sh1[t + off]; sh2[t] += sh2[t + off]; }
        __syncthreads();
    }
    if (t == 0) {
        atomicAdd(&sums[c], sh1[0]);
        atomicAdd(&sumsq[c], sh2[0]);
    }
}

__global__ void bn_stats(const float* __restrict__ sums, const float* __restrict__ sumsq,
                         float* __restrict__ mean, float* __restrict__ inv) {
    int c = threadIdx.x;
    const float cnt = (float)(NIMG * SPATIAL);
    float m = sums[c] / cnt;
    float v = sumsq[c] / cnt - m * m;
    mean[c] = m;
    inv[c] = rsqrtf(v + 1e-5f);
}

__global__ void bn_apply(float* __restrict__ out,
                         const float* __restrict__ mean, const float* __restrict__ inv) {
    int i = blockIdx.x * blockDim.x + threadIdx.x;
    int c = (i / (SPATIAL / 4)) % COUT;
    float4* p = (float4*)out;
    float4 v = p[i];
    float m = mean[c], iv = inv[c];
    v.x = (v.x - m) * iv;
    v.y = (v.y - m) * iv;
    v.z = (v.z - m) * iv;
    v.w = (v.w - m) * iv;
    p[i] = v;
}

extern "C" void kernel_launch(void* const* d_in, const int* in_sizes, int n_in,
                              void* d_out, int out_size, void* d_ws, size_t ws_size,
                              hipStream_t stream) {
    (void)in_sizes; (void)n_in; (void)out_size; (void)ws_size;
    const float* x    = (const float*)d_in[0];
    const float* w    = (const float*)d_in[1];
    const float* bias = (const float*)d_in[2];
    float* out = (float*)d_out;

    char* ws = (char*)d_ws;
    signed char* xpad = (signed char*)ws;
    signed char* wq   = (signed char*)(ws + XPAD_BYTES);
    float* stats      = (float*)(ws + XPAD_BYTES + WQ_BYTES);
    float* sums  = stats;
    float* sumsq = stats + 256;
    float* meanb = stats + 512;
    float* invb  = stats + 768;

    zero_ints<<<(XPAD_INTS + 255) / 256, 256, 0, stream>>>((int*)xpad, XPAD_INTS);
    zero_ints<<<2, 256, 0, stream>>>((int*)stats, 512);

    binarize_x_kernel<<<25088, 256, 0, stream>>>(x, (int*)xpad);
    binarize_w_kernel<<<2304, 256, 0, stream>>>(w, wq);

    conv_bnn_wmma<<<784, 256, 0, stream>>>(xpad, wq, bias, out);

    channel_reduce<<<dim3(256, 8), 256, 0, stream>>>(out, sums, sumsq);
    bn_stats<<<1, 256, 0, stream>>>(sums, sumsq, meanb, invb);
    bn_apply<<<25088, 256, 0, stream>>>(out, meanb, invb);
}
